// L2Attention_82351702934253
// MI455X (gfx1250) — matrix operand order repaired and checked by reading, not compiled, not run on hardware
//
#include <hip/hip_runtime.h>

// ---------------------------------------------------------------------------
// L2-distance attention for MI455X (gfx1250, wave32, WMMA).
//   qv = x @ Wqv^T ; q,v split/reshaped to [b,h,n,64]
//   attn = softmax_j( scale * (2*q_i.q_j - ||q_j||^2) )   (||q_i||^2 cancels)
//   out  = (attn @ v) reshaped @ Wout^T + b_out
// All matmuls on v_wmma_f32_16x16x32_f16 (f16 in, f32 accumulate).
// ---------------------------------------------------------------------------

typedef _Float16 half_t;
typedef __attribute__((ext_vector_type(16))) _Float16 v16h;
typedef __attribute__((ext_vector_type(8)))  _Float16 v8h;
typedef __attribute__((ext_vector_type(8)))  float    v8f;

constexpr int kBatch = 4;
constexpr int kSeq   = 2048;
constexpr int kDim   = 512;
constexpr int kHeads = 8;
constexpr int kDHead = 64;
constexpr int kInner = 512;               // kHeads * kDHead
constexpr int kRows  = kBatch * kSeq;     // 8192
constexpr float kScale = 0.125f;          // 64^-0.5

// ---- WMMA fragment loaders (layouts per CDNA5 ISA section 7.12.2) ----------
// A (16x32 f16, row-major source): lane holds row m=lane&15; element i of the
// v16h maps to k = (i>=8 ? 16 : 0) + 8*(lane>>4) + (i&7)  -> two 16B chunks.
__device__ __forceinline__ v16h load_a_frag(const half_t* base, int ld) {
  const int lane = threadIdx.x & 31;
  const int m    = lane & 15;
  const int hi   = lane >> 4;
  const half_t* row = base + (size_t)m * ld;
  v8h c0 = *(const v8h*)(row + hi * 8);
  v8h c1 = *(const v8h*)(row + 16 + hi * 8);
  v16h a;
#pragma unroll
  for (int i = 0; i < 8; ++i) { a[i] = c0[i]; a[i + 8] = c1[i]; }
  return a;
}

// B (32x16 f16), sourced as M2[n][k] (i.e. "N-major / transposed-B" memory):
// lane holds column n=lane&15; element i maps to k = (i>=8 ? 16 : 0) + 8*(lane>>4) + (i&7).
__device__ __forceinline__ v16h load_b_frag(const half_t* base, int ld) {
  const int lane = threadIdx.x & 31;
  const int n    = lane & 15;
  const int hi   = lane >> 4;
  const half_t* row = base + (size_t)n * ld;
  v8h c0 = *(const v8h*)(row + hi * 8);
  v8h c1 = *(const v8h*)(row + 16 + hi * 8);
  v16h b;
#pragma unroll
  for (int i = 0; i < 8; ++i) { b[i] = c0[i]; b[i + 8] = c1[i]; }
  return b;
}

__device__ __forceinline__ v8f wmma_f16(v16h a, v16h b, v8f c) {
  return __builtin_amdgcn_wmma_f32_16x16x32_f16(false, a, false, b,
                                                (short)0, c, false, false);
}

// C/D layout: acc[r] holds (row = r + 8*(lane>>4), col = lane&15).
// Row-wise reductions therefore reduce across each 16-lane half-group; xor
// masks < 16 never cross the group boundary on wave32.
__device__ __forceinline__ float grp16_max(float v) {
#pragma unroll
  for (int m = 1; m < 16; m <<= 1) v = fmaxf(v, __shfl_xor(v, m, 32));
  return v;
}
__device__ __forceinline__ float grp16_sum(float v) {
#pragma unroll
  for (int m = 1; m < 16; m <<= 1) v += __shfl_xor(v, m, 32);
  return v;
}

// ---------------------------------------------------------------------------
// Kernel 1: f32 -> f16 convert (grid-stride)
// ---------------------------------------------------------------------------
__global__ void cvt_f32_to_f16(const float* __restrict__ src,
                               half_t* __restrict__ dst, int n) {
  int i = blockIdx.x * blockDim.x + threadIdx.x;
  const int stride = gridDim.x * blockDim.x;
  for (; i < n; i += stride) dst[i] = (half_t)src[i];
}

// ---------------------------------------------------------------------------
// Kernel 2: qv = x_h @ Wqv_h^T.  Each wave: 16(M) x 64(N) tile; N tile == one
// head's 64 dims, so the q branch can also produce ||q||^2 per row.
// Outputs: qh [bh][n][64] f16, vt [bh][64][n] f16 (transposed), ql2 [bh][n] f32
// ---------------------------------------------------------------------------
__global__ void qv_proj_kernel(const half_t* __restrict__ xh,
                               const half_t* __restrict__ wqv,
                               half_t* __restrict__ qh,
                               half_t* __restrict__ vt,
                               float*  __restrict__ ql2) {
  const int wave = threadIdx.x >> 5;
  const int lane = threadIdx.x & 31;
  const int hi   = lane >> 4;
  const int mrow  = blockIdx.x * 64 + wave * 16;
  const int cbase = blockIdx.y * 64;

  v8f acc[4] = {};
  for (int kb = 0; kb < kDim; kb += 32) {
    v16h a = load_a_frag(xh + (size_t)mrow * kDim + kb, kDim);
    if (kb + 64 < kDim)
      __builtin_prefetch(xh + (size_t)mrow * kDim + kb + 64, 0, 1);
#pragma unroll
    for (int t = 0; t < 4; ++t) {
      v16h b = load_b_frag(wqv + (size_t)(cbase + t * 16) * kDim + kb, kDim);
      acc[t] = wmma_f16(a, b, acc[t]);
    }
  }

  const int bidx  = mrow >> 11;         // batch (rows 2048-aligned per batch)
  const int nbase = mrow & (kSeq - 1);
  if (cbase < kInner) {                 // ---- q + ||q||^2 ----
    const int head = cbase >> 6;
    const int bh   = bidx * kHeads + head;
    float sq[8];
#pragma unroll
    for (int r = 0; r < 8; ++r) sq[r] = 0.f;
#pragma unroll
    for (int t = 0; t < 4; ++t) {
      const int d = t * 16 + (lane & 15);
#pragma unroll
      for (int r = 0; r < 8; ++r) {
        const float qv = acc[t][r];
        sq[r] += qv * qv;
        const int nseq = nbase + r + 8 * hi;
        qh[((size_t)bh * kSeq + nseq) * kDHead + d] = (half_t)qv;
      }
    }
#pragma unroll
    for (int r = 0; r < 8; ++r) sq[r] = grp16_sum(sq[r]);
    if ((lane & 15) == 0) {             // lanes 0 and 16 cover rows 0-7 / 8-15
#pragma unroll
      for (int r = 0; r < 8; ++r)
        ql2[(size_t)bh * kSeq + nbase + r + 8 * hi] = sq[r];
    }
  } else {                              // ---- v (stored d-major) ----
    const int head = (cbase - kInner) >> 6;
    const int bh   = bidx * kHeads + head;
#pragma unroll
    for (int t = 0; t < 4; ++t) {
      const int d = t * 16 + (lane & 15);
#pragma unroll
      for (int r = 0; r < 8; ++r) {
        const int nseq = nbase + r + 8 * hi;
        vt[((size_t)bh * kDHead + d) * kSeq + nseq] = (half_t)acc[t][r];
      }
    }
  }
}

// ---------------------------------------------------------------------------
// Kernel 3: flash-style L2 attention. One wave per 16-query block, 32 keys per
// step: S(16x32) = Q(16x64).K^T via 4 WMMA, online softmax of
// scale*(2S - ||q_j||^2), P re-laid out D->A through a wave-private LDS slab,
// O(16x64) += P.V via 4 WMMA.  Output oh [b][n][512] f16.
// ---------------------------------------------------------------------------
__global__ void l2_flash_kernel(const half_t* __restrict__ qh,
                                const half_t* __restrict__ vt,
                                const float*  __restrict__ ql2,
                                half_t* __restrict__ oh) {
  __shared__ __align__(16) half_t pbuf[4 * 16 * 32];
  const int wave = threadIdx.x >> 5;
  const int lane = threadIdx.x & 31;
  const int hi   = lane >> 4;
  const int bh    = blockIdx.y;
  const int qbase = (blockIdx.x * 4 + wave) * 16;

  const half_t* qbh  = qh  + (size_t)bh * kSeq * kDHead;
  const half_t* vbh  = vt  + (size_t)bh * kDHead * kSeq;
  const float*  l2bh = ql2 + (size_t)bh * kSeq;
  half_t* slab = pbuf + wave * (16 * 32);

  const v16h aq0 = load_a_frag(qbh + (size_t)qbase * kDHead + 0,  kDHead);
  const v16h aq1 = load_a_frag(qbh + (size_t)qbase * kDHead + 32, kDHead);

  float mstat[8], lstat[8];
  v8f o[4] = {};
#pragma unroll
  for (int r = 0; r < 8; ++r) { mstat[r] = -__builtin_inff(); lstat[r] = 0.f; }

  for (int kb = 0; kb < kSeq; kb += 32) {
    const half_t* krow = qbh + (size_t)kb * kDHead;   // keys == q rows
    v16h b00 = load_b_frag(krow, kDHead);
    v16h b01 = load_b_frag(krow + 32, kDHead);
    v16h b10 = load_b_frag(krow + 16 * kDHead, kDHead);
    v16h b11 = load_b_frag(krow + 16 * kDHead + 32, kDHead);
    v8f s0 = {}, s1 = {};
    s0 = wmma_f16(aq0, b00, s0); s0 = wmma_f16(aq1, b01, s0);
    s1 = wmma_f16(aq0, b10, s1); s1 = wmma_f16(aq1, b11, s1);

    const float jl0 = l2bh[kb + (lane & 15)];
    const float jl1 = l2bh[kb + 16 + (lane & 15)];
    float z0[8], z1[8], alpha[8], p0[8], p1[8];
#pragma unroll
    for (int r = 0; r < 8; ++r) {
      z0[r] = kScale * (2.f * s0[r] - jl0);
      z1[r] = kScale * (2.f * s1[r] - jl1);
    }
#pragma unroll
    for (int r = 0; r < 8; ++r) {
      const float mnew = fmaxf(mstat[r], grp16_max(fmaxf(z0[r], z1[r])));
      alpha[r] = __expf(mstat[r] - mnew);
      p0[r] = __expf(z0[r] - mnew);
      p1[r] = __expf(z1[r] - mnew);
      lstat[r] = lstat[r] * alpha[r] + grp16_sum(p0[r] + p1[r]);
      mstat[r] = mnew;
    }
#pragma unroll
    for (int t = 0; t < 4; ++t)
#pragma unroll
      for (int r = 0; r < 8; ++r) o[t][r] *= alpha[r];

    // P: D-layout regs -> LDS (row-major 16x32) -> A-fragment.
    __syncthreads();
#pragma unroll
    for (int r = 0; r < 8; ++r) {
      const int prow = r + 8 * hi;
      slab[prow * 32 + (lane & 15)]      = (half_t)p0[r];
      slab[prow * 32 + 16 + (lane & 15)] = (half_t)p1[r];
    }
    __syncthreads();
    const v16h ap = load_a_frag(slab, 32);
#pragma unroll
    for (int t = 0; t < 4; ++t) {
      v16h bv = load_b_frag(vbh + (size_t)(t * 16) * kSeq + kb, kSeq);
      o[t] = wmma_f16(ap, bv, o[t]);
    }
  }

  const int b = bh >> 3, head = bh & 7;
#pragma unroll
  for (int t = 0; t < 4; ++t) {
    const int col = head * kDHead + t * 16 + (lane & 15);
#pragma unroll
    for (int r = 0; r < 8; ++r) {
      const int nseq = qbase + r + 8 * hi;
      oh[((size_t)b * kSeq + nseq) * kInner + col] = (half_t)(o[t][r] / lstat[r]);
    }
  }
}

// ---------------------------------------------------------------------------
// Kernel 4: out = oh @ Wout^T + b_out  (f32 stores)
// ---------------------------------------------------------------------------
__global__ void out_proj_kernel(const half_t* __restrict__ oh,
                                const half_t* __restrict__ wout,
                                const float*  __restrict__ bout,
                                float* __restrict__ out) {
  const int wave = threadIdx.x >> 5;
  const int lane = threadIdx.x & 31;
  const int hi   = lane >> 4;
  const int mrow  = blockIdx.x * 64 + wave * 16;
  const int cbase = blockIdx.y * 64;

  v8f acc[4] = {};
  for (int kb = 0; kb < kInner; kb += 32) {
    v16h a = load_a_frag(oh + (size_t)mrow * kInner + kb, kInner);
    if (kb + 64 < kInner)
      __builtin_prefetch(oh + (size_t)mrow * kInner + kb + 64, 0, 1);
#pragma unroll
    for (int t = 0; t < 4; ++t) {
      v16h b = load_b_frag(wout + (size_t)(cbase + t * 16) * kInner + kb, kInner);
      acc[t] = wmma_f16(a, b, acc[t]);
    }
  }
#pragma unroll
  for (int t = 0; t < 4; ++t) {
    const int col = cbase + t * 16 + (lane & 15);
    const float bias = bout[col];
#pragma unroll
    for (int r = 0; r < 8; ++r) {
      const int row = mrow + r + 8 * hi;
      out[(size_t)row * kDim + col] = acc[t][r] + bias;
    }
  }
}

// ---------------------------------------------------------------------------
extern "C" void kernel_launch(void* const* d_in, const int* in_sizes, int n_in,
                              void* d_out, int out_size, void* d_ws, size_t ws_size,
                              hipStream_t stream) {
  const float* x    = (const float*)d_in[0];   // [4,2048,512]
  const float* Wqv  = (const float*)d_in[1];   // [1024,512]
  const float* Wout = (const float*)d_in[2];   // [512,512]
  const float* bout = (const float*)d_in[3];   // [512]
  float* out = (float*)d_out;                  // [4,2048,512] f32

  char* ws = (char*)d_ws;
  size_t off = 0;
  auto carve = [&](size_t bytes) -> void* {
    void* p = ws + off;
    off = (off + bytes + 255) & ~(size_t)255;
    return p;
  };
  half_t* xh    = (half_t*)carve((size_t)kRows * kDim * sizeof(half_t));          // 8 MB
  half_t* wqvh  = (half_t*)carve((size_t)2 * kInner * kDim * sizeof(half_t));     // 1 MB
  half_t* wouth = (half_t*)carve((size_t)kDim * kInner * sizeof(half_t));         // 0.5 MB
  half_t* qh    = (half_t*)carve((size_t)kBatch * kHeads * kSeq * kDHead * sizeof(half_t)); // 8 MB
  half_t* vt    = (half_t*)carve((size_t)kBatch * kHeads * kDHead * kSeq * sizeof(half_t)); // 8 MB
  float*  ql2   = (float*) carve((size_t)kBatch * kHeads * kSeq * sizeof(float)); // 0.25 MB
  // x_h is fully consumed by qv_proj_kernel before l2_flash_kernel runs on the
  // same stream, so reuse its slab for the pre-projection attention output.
  half_t* oh = xh;

  cvt_f32_to_f16<<<dim3(512), dim3(256), 0, stream>>>(x, xh, kRows * kDim);
  cvt_f32_to_f16<<<dim3(128), dim3(256), 0, stream>>>(Wqv, wqvh, 2 * kInner * kDim);
  cvt_f32_to_f16<<<dim3(64),  dim3(256), 0, stream>>>(Wout, wouth, kDim * kInner);

  qv_proj_kernel<<<dim3(kRows / 64, (2 * kInner) / 64), dim3(128), 0, stream>>>(
      xh, wqvh, qh, vt, ql2);

  l2_flash_kernel<<<dim3(kSeq / 64, kBatch * kHeads), dim3(128), 0, stream>>>(
      qh, vt, ql2, oh);

  out_proj_kernel<<<dim3(kRows / 64, kDim / 64), dim3(128), 0, stream>>>(
      oh, wouth, bout, out);
}
